// MultiHeadAttention_46497315946956
// MI455X (gfx1250) — matrix operand order repaired and checked by reading, not compiled, not run on hardware
//
#include <hip/hip_runtime.h>
#include <hip/hip_bf16.h>

typedef __attribute__((ext_vector_type(16))) __bf16       v16bf;
typedef __attribute__((ext_vector_type(8)))  float        v8f;
typedef __attribute__((ext_vector_type(8)))  unsigned int v8u;

__device__ __forceinline__ unsigned short f2bf(float f) {
    unsigned int u = __builtin_bit_cast(unsigned int, f);
    unsigned int r = u + 0x7FFFu + ((u >> 16) & 1u);   // round-to-nearest-even
    return (unsigned short)(r >> 16);
}
__device__ __forceinline__ unsigned int f2bf2(float lo, float hi) {
    return (unsigned int)f2bf(lo) | ((unsigned int)f2bf(hi) << 16);
}

// CDNA5 async copy: global -> LDS, 16 bytes per lane, tracked with ASYNCcnt.
__device__ __forceinline__ void async_ld_b128(unsigned lds_byte_addr, const void* gaddr) {
    asm volatile("global_load_async_to_lds_b128 %0, %1, off"
                 :: "v"(lds_byte_addr), "v"(gaddr) : "memory");
}
__device__ __forceinline__ void wait_async0() {
    asm volatile("s_wait_asynccnt 0" ::: "memory");
}
__device__ __forceinline__ void wait_ds0() {
    asm volatile("s_wait_dscnt 0" ::: "memory");
}

// ---------------------------------------------------------------------------
// WMMA GEMM:  Out[M,N] = A[M,K] @ W[N,K]^T + bias
//   A_BF16=false : A fp32 (converted to bf16 on LDS store)
//   A_BF16=true  : A bf16 (staged via global_load_async_to_lds_b128)
//   OUT_F32      : fp32 out vs bf16 out
// Block tile 128x128, K-step 32, 256 threads = 8 waves, each wave 32x64
// (2 A-frags x 4 B-frags = 8 WMMAs per K-step per wave).
// ---------------------------------------------------------------------------
template<bool A_BF16, bool OUT_F32>
__global__ __launch_bounds__(256) void gemm_wmma_kernel(
    const void* __restrict__ Aptr, const float* __restrict__ W,
    const float* __restrict__ bias, void* __restrict__ Out,
    int M, int N, int K)
{
    constexpr int AS_STRIDE = A_BF16 ? 32 : 36;   // bf16 path: 64B rows, 16B aligned for async
    constexpr int AS_ELEMS  = 128 * AS_STRIDE;
    constexpr int BT_STRIDE = 136;                // [k][n] transposed W tile, padded
    __shared__ unsigned short smem[AS_ELEMS + 32 * BT_STRIDE];
#define AS_(m, k) smem[(m) * AS_STRIDE + (k)]
#define BT_(k, n) smem[AS_ELEMS + (k) * BT_STRIDE + (n)]

    const int tid  = threadIdx.x;
    const int lane = tid & 31;
    const int wave = tid >> 5;
    const int mBlock = blockIdx.y * 128;
    const int nBlock = blockIdx.x * 128;
    const int wm = (wave >> 1) * 32;   // 4 row-strips of 32
    const int wn = (wave & 1) * 64;    // 2 col-strips of 64

    const int am = lane & 15;          // A-fragment row
    const int kh = (lane >> 4) * 8;    // A-fragment K half offset

    v8f acc[2][4] = {};

    for (int k0 = 0; k0 < K; k0 += 32) {
        __syncthreads();
        // ---- stage A tile (128 x 32) ----
        if constexpr (A_BF16) {
            // 512 x 16B chunks, 2 per thread, async DMA into LDS
            #pragma unroll
            for (int i = 0; i < 2; ++i) {
                int linear = tid + i * 256;          // chunk id
                int row = linear >> 2;               // 4 chunks per row
                int c   = linear & 3;
                const char* g = (const char*)Aptr +
                    (((size_t)(mBlock + row) * K + k0 + c * 8) << 1);
                async_ld_b128((unsigned)((row * AS_STRIDE + c * 8) << 1), g);
            }
        } else {
            const float4* A4 = (const float4*)Aptr;
            #pragma unroll
            for (int i = 0; i < 4; ++i) {
                int linear = tid + i * 256;          // float4 units, 1024 total
                int row = linear >> 3;               // 8 float4 per row
                int kc  = (linear & 7) * 4;
                float4 f = A4[((size_t)(mBlock + row) * K + k0 + kc) >> 2];
                *(unsigned int*)&AS_(row, kc)     = f2bf2(f.x, f.y);
                *(unsigned int*)&AS_(row, kc + 2) = f2bf2(f.z, f.w);
            }
        }
        // ---- stage W tile (128 x 32) transposed -> BT_[k][n], bf16 ----
        {
            const float4* W4 = (const float4*)W;
            #pragma unroll
            for (int i = 0; i < 4; ++i) {
                int linear = tid + i * 256;          // float4 units, 1024 total
                int n  = linear >> 3;
                int kc = (linear & 7) * 4;
                float4 f = W4[((size_t)(nBlock + n) * K + k0 + kc) >> 2];
                BT_(kc, n)     = f2bf(f.x);
                BT_(kc + 1, n) = f2bf(f.y);
                BT_(kc + 2, n) = f2bf(f.z);
                BT_(kc + 3, n) = f2bf(f.w);
            }
        }
        if constexpr (A_BF16) wait_async0();
        __syncthreads();

        // ---- fragments + 8 WMMAs per wave ----
        v16bf a[2];
        #pragma unroll
        for (int i = 0; i < 2; ++i) {
            int mrow = wm + i * 16 + am;
            v8u r;
            #pragma unroll
            for (int v = 0; v < 4; ++v) r[v]     = *(const unsigned int*)&AS_(mrow, kh + 2 * v);
            #pragma unroll
            for (int v = 0; v < 4; ++v) r[4 + v] = *(const unsigned int*)&AS_(mrow, 16 + kh + 2 * v);
            a[i] = __builtin_bit_cast(v16bf, r);
        }
        #pragma unroll
        for (int j = 0; j < 4; ++j) {
            v8u r;
            #pragma unroll
            for (int v = 0; v < 8; ++v) {
                int kk = (v < 4) ? kh + 2 * v : 16 + kh + 2 * (v - 4);
                r[v] = (unsigned int)BT_(kk, wn + j * 16 + am) | ((unsigned int)BT_(kk + 1, wn + j * 16 + am) << 16);
            }
            v16bf b = __builtin_bit_cast(v16bf, r);
            #pragma unroll
            for (int i = 0; i < 2; ++i)
                acc[i][j] = __builtin_amdgcn_wmma_f32_16x16x32_bf16(
                    false, a[i], false, b, (short)0, acc[i][j], false, false);
        }
    }

    // ---- epilogue: bias + store ----
    #pragma unroll
    for (int j = 0; j < 4; ++j) {
        int col = nBlock + wn + j * 16 + (lane & 15);
        float bj = bias[col];
        #pragma unroll
        for (int i = 0; i < 2; ++i) {
            #pragma unroll
            for (int r = 0; r < 8; ++r) {
                int row = mBlock + wm + i * 16 + r + ((lane >> 4) << 3);
                float val = acc[i][j][r] + bj;
                if constexpr (OUT_F32)
                    ((float*)Out)[(size_t)row * N + col] = val;
                else
                    ((unsigned short*)Out)[(size_t)row * N + col] = f2bf(val);
            }
        }
    }
#undef AS_
#undef BT_
}

// ---------------------------------------------------------------------------
// Flash-attention (causal). Workgroup = 4 waves x 16 q-rows of one (b,h).
// Streams 32-key blocks: K transposed by VALU into LDS, V via async DMA.
// ---------------------------------------------------------------------------
#define ATT_DK 64
__global__ __launch_bounds__(128) void attn_flash_kernel(
    const unsigned short* __restrict__ Q, const unsigned short* __restrict__ Kb,
    const unsigned short* __restrict__ Vb, unsigned short* __restrict__ Ctx,
    int S, int D, int H)
{
    // single LDS block so async byte addresses are known (group segment base 0)
    // Kt: [d][key] 64x36 @0 ; Vt: [key][d] 32x64 @2304 ; Pw: 4x16x36 @4352
    __shared__ unsigned short smem[64 * 36 + 32 * 64 + 4 * 16 * 36];
#define KT_(d, k)    smem[(d) * 36 + (k)]
#define VT_(key, d)  smem[2304 + (key) * 64 + (d)]
#define PW_(w, q, k) smem[4352 + (w) * 576 + (q) * 36 + (k)]
    constexpr unsigned VT_BYTE = 2304u * 2u;   // 16B aligned

    const int qBlk = blockIdx.x * 64;
    const int bh = blockIdx.y;
    const int b = bh / H, h = bh % H;
    const int tid = threadIdx.x, lane = tid & 31, wave = tid >> 5;
    const int qw = qBlk + wave * 16;           // first q row of this wave
    const int qwu = __builtin_amdgcn_readfirstlane(qw);

    const int am = lane & 15;                  // A-frag row within tile
    const int kh = (lane >> 4) * 8;            // A-frag K half offset
    const int rrow = (lane >> 4) << 3;         // C-layout row offset per lane half

    // ---- resident Q A-fragments: two K-steps of 32 over DK=64 ----
    v16bf qa[2];
    #pragma unroll
    for (int s2 = 0; s2 < 2; ++s2) {
        const unsigned short* qrow =
            Q + ((size_t)(b * S + qw + am)) * D + h * ATT_DK + s2 * 32;
        v8u r;
        #pragma unroll
        for (int v = 0; v < 4; ++v) r[v]     = *(const unsigned int*)(qrow + kh + 2 * v);
        #pragma unroll
        for (int v = 0; v < 4; ++v) r[4 + v] = *(const unsigned int*)(qrow + 16 + kh + 2 * v);
        qa[s2] = __builtin_bit_cast(v16bf, r);
    }

    v8f o[4] = {};                  // context accumulators, 4 d-tiles of 16
    float Mr[8], Lr[8];
    #pragma unroll
    for (int r = 0; r < 8; ++r) { Mr[r] = -1e30f; Lr[r] = 0.f; }

    const int nkb = qBlk / 32 + 2;  // key blocks covering keys <= qBlk+63

    for (int kb = 0; kb < nkb; ++kb) {
        __syncthreads();
        // ---- stage K (transposed via VALU) and V (async DMA) ----
        {
            int key = tid >> 2;                 // 32 keys, 4 threads per key
            int dc  = (tid & 3) * 16;           // 16 d-elements per thread
            const unsigned short* ks =
                Kb + ((size_t)(b * S + kb * 32 + key)) * D + h * ATT_DK + dc;
            #pragma unroll
            for (int i = 0; i < 8; ++i) {
                unsigned int u = *(const unsigned int*)(ks + 2 * i);
                KT_(dc + 2 * i, key)     = (unsigned short)(u & 0xFFFFu);
                KT_(dc + 2 * i + 1, key) = (unsigned short)(u >> 16);
            }
            // V: 256 x 16B chunks, 2 per thread
            #pragma unroll
            for (int i = 0; i < 2; ++i) {
                int c    = tid + i * 128;       // chunk id
                int vkey = c >> 3;              // 8 chunks per 128B row
                int dch  = c & 7;
                const char* g = (const char*)(Vb +
                    ((size_t)(b * S + kb * 32 + vkey)) * D + h * ATT_DK + dch * 8);
                async_ld_b128(VT_BYTE + (unsigned)(vkey * 128 + dch * 16), g);
            }
            if (kb + 1 < nkb) {                 // prefetch next key block
                __builtin_prefetch(ks + (size_t)32 * D, 0, 1);
                __builtin_prefetch(Vb + ((size_t)(b * S + (kb + 1) * 32 + key)) * D
                                      + h * ATT_DK + dc, 0, 1);
            }
        }
        wait_async0();
        __syncthreads();

        if (kb * 32 > qwu + 15) continue;       // block fully masked (scalar branch)

        // ---- scores: S = Q K^T (two 16x16 C tiles over 32 keys) ----
        v8f sc[2] = {};
        #pragma unroll
        for (int s2 = 0; s2 < 2; ++s2) {
            #pragma unroll
            for (int t = 0; t < 2; ++t) {
                v8u r;
                #pragma unroll
                for (int v = 0; v < 8; ++v) {
                    int kk = s2 * 32 + ((v < 4) ? kh + 2 * v : 16 + kh + 2 * (v - 4));
                    r[v] = (unsigned int)KT_(kk, t * 16 + am) | ((unsigned int)KT_(kk + 1, t * 16 + am) << 16);
                }
                v16bf bk = __builtin_bit_cast(v16bf, r);
                sc[t] = __builtin_amdgcn_wmma_f32_16x16x32_bf16(
                    false, qa[s2], false, bk, (short)0, sc[t], false, false);
            }
        }

        // ---- online softmax (rows live across 16-lane groups) ----
        const float scale = 0.125f;             // 1/sqrt(64)
        #pragma unroll
        for (int r = 0; r < 8; ++r) {
            int q = qw + rrow + r;
            float mx = -1e30f;
            #pragma unroll
            for (int t = 0; t < 2; ++t) {
                int key = kb * 32 + t * 16 + (lane & 15);
                float v = sc[t][r] * scale;
                if (key > q) v = -1e30f;        // causal mask
                sc[t][r] = v;
                mx = fmaxf(mx, v);
            }
            #pragma unroll
            for (int off = 1; off < 16; off <<= 1)
                mx = fmaxf(mx, __shfl_xor(mx, off, 32));
            float mnew  = fmaxf(Mr[r], mx);
            float alpha = __expf(Mr[r] - mnew);
            Mr[r] = mnew;
            float rsum = 0.f;
            #pragma unroll
            for (int t = 0; t < 2; ++t) {
                float p = __expf(sc[t][r] - mnew);
                sc[t][r] = p;
                rsum += p;
            }
            #pragma unroll
            for (int off = 1; off < 16; off <<= 1)
                rsum += __shfl_xor(rsum, off, 32);
            Lr[r] = Lr[r] * alpha + rsum;
            #pragma unroll
            for (int t = 0; t < 4; ++t) o[t][r] *= alpha;
        }

        // ---- P (C layout) -> LDS -> A fragment ----
        #pragma unroll
        for (int t = 0; t < 2; ++t)
            #pragma unroll
            for (int r = 0; r < 8; ++r)
                PW_(wave, rrow + r, t * 16 + (lane & 15)) = f2bf(sc[t][r]);
        wait_ds0();

        v8u pr;
        #pragma unroll
        for (int v = 0; v < 4; ++v) pr[v]     = *(const unsigned int*)&PW_(wave, am, kh + 2 * v);
        #pragma unroll
        for (int v = 0; v < 4; ++v) pr[4 + v] = *(const unsigned int*)&PW_(wave, am, 16 + kh + 2 * v);
        v16bf pa = __builtin_bit_cast(v16bf, pr);

        // ---- context: O += P @ V (4 d-tiles of 16) ----
        #pragma unroll
        for (int t = 0; t < 4; ++t) {
            v8u r;
            #pragma unroll
            for (int v = 0; v < 8; ++v) {
                int kk = (v < 4) ? kh + 2 * v : 16 + kh + 2 * (v - 4);
                r[v] = (unsigned int)VT_(kk, t * 16 + am) | ((unsigned int)VT_(kk + 1, t * 16 + am) << 16);
            }
            v16bf bv = __builtin_bit_cast(v16bf, r);
            o[t] = __builtin_amdgcn_wmma_f32_16x16x32_bf16(
                false, pa, false, bv, (short)0, o[t], false, false);
        }
    }

    // ---- normalize + store context (bf16) ----
    #pragma unroll
    for (int r = 0; r < 8; ++r) {
        float inv = 1.f / Lr[r];
        #pragma unroll
        for (int t = 0; t < 4; ++t) o[t][r] *= inv;
    }
    #pragma unroll
    for (int t = 0; t < 4; ++t)
        #pragma unroll
        for (int r = 0; r < 8; ++r) {
            int q = qw + rrow + r;
            int d = t * 16 + (lane & 15);
            Ctx[((size_t)(b * S + q)) * D + h * ATT_DK + d] = f2bf(o[t][r]);
        }
#undef KT_
#undef VT_
#undef PW_
}

// ---------------------------------------------------------------------------
extern "C" void kernel_launch(void* const* d_in, const int* in_sizes, int n_in,
                              void* d_out, int out_size, void* d_ws, size_t ws_size,
                              hipStream_t stream)
{
    (void)in_sizes; (void)n_in; (void)out_size; (void)ws_size;
    const int B = 2, S = 2048, D = 1024, H = 16;
    const int M = B * S, N = D, K = D;

    // workspace: Q, K, V, context as bf16 (8 MiB each)
    unsigned short* Qbf = (unsigned short*)d_ws;
    unsigned short* Kbf = Qbf + (size_t)M * D;
    unsigned short* Vbf = Kbf + (size_t)M * D;
    unsigned short* Cbf = Vbf + (size_t)M * D;

    dim3 gg(N / 128, M / 128), gb(256);

    // inputs: 0 query, 1 key, 2 value, 3 mask(causal tril, handled in-kernel),
    //         4 wq, 5 bq, 6 wk, 7 bk, 8 wv, 9 bv, 10 wo, 11 bo
    gemm_wmma_kernel<false, false><<<gg, gb, 0, stream>>>(
        d_in[0], (const float*)d_in[4], (const float*)d_in[5], Qbf, M, N, K);
    gemm_wmma_kernel<false, false><<<gg, gb, 0, stream>>>(
        d_in[1], (const float*)d_in[6], (const float*)d_in[7], Kbf, M, N, K);
    gemm_wmma_kernel<false, false><<<gg, gb, 0, stream>>>(
        d_in[2], (const float*)d_in[8], (const float*)d_in[9], Vbf, M, N, K);

    attn_flash_kernel<<<dim3(S / 64, B * H), 128, 0, stream>>>(
        Qbf, Kbf, Vbf, Cbf, S, D, H);

    gemm_wmma_kernel<true, true><<<gg, gb, 0, stream>>>(
        Cbf, (const float*)d_in[10], (const float*)d_in[11], d_out, M, N, K);
}
